// MHA_Res2proj_88210038326177
// MI455X (gfx1250) — matrix operand order repaired and checked by reading, not compiled, not run on hardware
//
#include <hip/hip_runtime.h>

// ---------------- constants ----------------
constexpr int K_B  = 2;
constexpr int K_T  = 2304;      // 48*48
constexpr int K_H  = 48;
constexpr int K_W  = 48;
constexpr int K_C  = 256;
constexpr int K_C4 = 64;
constexpr int K_HEADS = 4;
constexpr int K_DH = 64;
constexpr int K_SBH = 4 * K_B * K_HEADS;   // 32

// ---------------- WMMA types / helpers ----------------
typedef __attribute__((ext_vector_type(16))) __bf16 v16bf;
typedef __attribute__((ext_vector_type(8)))  float  v8f;

struct FragBF {
    union { v16bf v; unsigned short u[16]; uint4 q[2]; };
};

static __device__ __forceinline__ unsigned short f2bf(float f) {
    union { float f; unsigned u; } c; c.f = f;
    unsigned r = c.u + 0x7FFFu + ((c.u >> 16) & 1u);   // round-to-nearest-even
    return (unsigned short)(r >> 16);
}

// A-matrix 16x32 bf16 fragment (ISA 7.12.2): lane m=lane&15, half=lane>>4,
// element e -> k = (e>=8?16:0) + half*8 + (e&7). Two contiguous uint4 loads.
static __device__ __forceinline__ void load_fragA(FragBF& f, const unsigned short* src,
                                                  int stride, int lane) {
    int m = lane & 15, half = lane >> 4;
    const unsigned short* p = src + (size_t)m * stride + half * 8;
    f.q[0] = *(const uint4*)(p);
    f.q[1] = *(const uint4*)(p + 16);
}

// B-matrix 32x16 bf16 fragment: lane n=lane&15 selects B column n
// (stored as row n of `src`), element e -> k = (e>=8?16:0) + half*8 + (e&7).
static __device__ __forceinline__ void load_fragB(FragBF& f, const unsigned short* src,
                                                  int stride, int lane) {
    int n = lane & 15, half = lane >> 4;
    const unsigned short* p = src + (size_t)n * stride + half * 8;
    f.q[0] = *(const uint4*)(p);
    f.q[1] = *(const uint4*)(p + 16);
}

static __device__ __forceinline__ v8f wmma_bf16(const FragBF& a, const FragBF& b, v8f c) {
    return __builtin_amdgcn_wmma_f32_16x16x32_bf16(false, a.v, false, b.v,
                                                   (short)0, c, false, false);
}

// ---------------- 1) conv_in (1x1, 256->256) + BN1 -> identity[b][c][t] ----------------
__global__ void k_convin(const float* __restrict__ x, const float* __restrict__ w,
                         const float* __restrict__ g, const float* __restrict__ bb,
                         const float* __restrict__ mm, const float* __restrict__ vv,
                         float* __restrict__ idn) {
    int o = blockIdx.x & 255;
    int b = blockIdx.x >> 8;
    int t = blockIdx.y * 256 + threadIdx.x;
    const float* xr = x + ((size_t)b * K_T + t) * K_C;
    const float* wr = w + (size_t)o * K_C;
    float acc = 0.f;
#pragma unroll 8
    for (int c = 0; c < K_C; ++c) acc += xr[c] * wr[c];
    float inv = g[o] * rsqrtf(vv[o] + 1e-5f);
    idn[((size_t)b * K_C + o) * K_T + t] = acc * inv + (bb[o] - mm[o] * inv);
}

// ---------------- 2) Res2Net scale: dw3x3 + pw(64x64) + BN2 ----------------
__global__ void k_res2(const float* __restrict__ idn, const float* __restrict__ yprev,
                       const float* __restrict__ dw, const float* __restrict__ pw,
                       const float* __restrict__ g, const float* __restrict__ bb,
                       const float* __restrict__ mm, const float* __restrict__ vv,
                       float* __restrict__ yout, int chan_off, int add_prev) {
    __shared__ float z[K_C4];
    int bt = blockIdx.x;
    int b = bt / K_T, t = bt % K_T;
    int hh = t / K_W, ww = t % K_W;
    int c = threadIdx.x;

    float acc = 0.f;
    const float* src = idn + ((size_t)b * K_C + chan_off + c) * K_T;
    const float* ps  = yprev + ((size_t)b * K_C4 + c) * K_T;
#pragma unroll
    for (int ky = 0; ky < 3; ++ky) {
        int y_ = hh + ky - 1;
        if (y_ < 0 || y_ >= K_H) continue;
#pragma unroll
        for (int kx = 0; kx < 3; ++kx) {
            int x_ = ww + kx - 1;
            if (x_ < 0 || x_ >= K_W) continue;
            int tt = y_ * K_W + x_;
            float val = src[tt];
            if (add_prev) val += ps[tt];
            acc += val * dw[c * 9 + ky * 3 + kx];
        }
    }
    z[c] = acc;
    __syncthreads();
    int o = threadIdx.x;
    float s2 = 0.f;
#pragma unroll 8
    for (int cc = 0; cc < K_C4; ++cc) s2 += z[cc] * pw[o * K_C4 + cc];
    float inv = g[o] * rsqrtf(vv[o] + 1e-5f);
    yout[((size_t)b * K_C4 + o) * K_T + t] = s2 * inv + (bb[o] - mm[o] * inv);
}

// ---------------- 3) build r[s][b][t][c] (bf16) ----------------
__global__ void k_buildr(const float* __restrict__ idn, const float* __restrict__ ys,
                         unsigned short* __restrict__ r) {
    size_t i = (size_t)blockIdx.x * 256 + threadIdx.x;    // 4*B*T*64 total
    int c = (int)(i & 63);
    size_t j = i >> 6;
    int t = (int)(j % K_T); j /= K_T;
    int b = (int)(j % K_B);
    int s = (int)(j / K_B);
    float val;
    if (s == 0) val = idn[((size_t)b * K_C + c) * K_T + t];
    else        val = ys[(((size_t)(s - 1) * K_B + b) * K_C4 + c) * K_T + t];
    r[i] = f2bf(val);
}

// ---------------- weight f32 -> bf16 ----------------
__global__ void k_cvt(const float* __restrict__ src, unsigned short* __restrict__ dst, int n) {
    int i = blockIdx.x * 256 + threadIdx.x;
    if (i < n) dst[i] = f2bf(src[i]);
}

// ---------------- 4) qkv projection (WMMA, K=64) ----------------
// grid: (s*B, tile-groups, proj). block 128 = 4 waves, one 16x16 tile each.
__global__ __launch_bounds__(128) void k_qkv(const unsigned short* __restrict__ r,
                                             const unsigned short* __restrict__ WB,
                                             unsigned short* __restrict__ Q,
                                             unsigned short* __restrict__ Kk,
                                             unsigned short* __restrict__ V) {
    int sb   = blockIdx.x;                           // s*B + b
    int wave = threadIdx.x >> 5, lane = threadIdx.x & 31;
    int tile = blockIdx.y * 4 + wave;                // 0..2303
    int trow = tile >> 4;                            // 0..143  (T tiles)
    int tcol = tile & 15;                            // 0..15   (N tiles, N=256)
    int proj = blockIdx.z;

    const unsigned short* A  = r  + ((size_t)sb * K_T + (size_t)trow * 16) * K_C4;
    const unsigned short* Wb = WB + (size_t)proj * 256 * 64 + (size_t)tcol * 16 * 64;

    v8f acc = {};
    FragBF a, b;
    load_fragA(a, A, K_C4, lane);        load_fragB(b, Wb, 64, lane);
    acc = wmma_bf16(a, b, acc);
    load_fragA(a, A + 32, K_C4, lane);   load_fragB(b, Wb + 32, 64, lane);
    acc = wmma_bf16(a, b, acc);

    unsigned short* outp = (proj == 0) ? Q : (proj == 1) ? Kk : V;
    int n = lane & 15, half = lane >> 4;
    int o = tcol * 16 + n;
    int h = o >> 6, d = o & 63;
    size_t base = ((size_t)(sb * K_HEADS + h) * K_T) * K_DH + d;
#pragma unroll
    for (int v = 0; v < 8; ++v) {
        int t = trow * 16 + v + 8 * half;            // C/D layout: row = v + 8*half
        outp[base + (size_t)t * K_DH] = f2bf(acc[v]);
    }
}

// ---------------- 5) flash attention (WMMA + async global->LDS K staging) ----------------
// grid: (32 sbh, T/64). block 128 = 4 waves; each wave owns 16 query rows.
__global__ __launch_bounds__(128) void k_attn(const unsigned short* __restrict__ Q,
                                              const unsigned short* __restrict__ Kk,
                                              const unsigned short* __restrict__ V,
                                              unsigned short* __restrict__ O) {
    __shared__ __align__(16) unsigned short sK[32 * 72];        // K tile, stride 72
    __shared__ __align__(16) unsigned short sVt[64 * 40];       // V tile transposed [d][t]
    __shared__ __align__(16) unsigned short sP[4][16 * 40];     // per-wave P tile

    int sbh  = blockIdx.x;
    int wave = threadIdx.x >> 5, lane = threadIdx.x & 31;
    int q0   = blockIdx.y * 64 + wave * 16;
    int s = sbh >> 3, b = (sbh >> 2) & 1, h = sbh & 3;          // sbh = s*8 + b*4 + h

    const unsigned short* Qb = Q  + (size_t)sbh * K_T * K_DH;
    const unsigned short* Kb = Kk + (size_t)sbh * K_T * K_DH;
    const unsigned short* Vb = V  + (size_t)sbh * K_T * K_DH;

    FragBF aq0, aq1;
    load_fragA(aq0, Qb + (size_t)q0 * K_DH,      K_DH, lane);
    load_fragA(aq1, Qb + (size_t)q0 * K_DH + 32, K_DH, lane);

    v8f o0 = {}, o1 = {}, o2 = {}, o3 = {};
    float mrow[8], lrow[8];
#pragma unroll
    for (int v = 0; v < 8; ++v) { mrow[v] = -1e30f; lrow[v] = 0.f; }
    int n = lane & 15, half = lane >> 4;

    // per-thread staging coords: row r_ = tid/4, 16-element column chunk c_
    int r_ = threadIdx.x >> 2;
    int c_ = (threadIdx.x & 3) * 16;
    // LDS byte address of this thread's K-tile destination (aperture rule: addr[31:0])
    unsigned ldsK = (unsigned)(size_t)(sK + r_ * 72 + c_);

    for (int j0 = 0; j0 < K_T; j0 += 32) {
        // --- K tile: async copy global -> LDS (no VGPR roundtrip, ASYNCcnt) ---
        {
            const unsigned short* gk = Kb + (size_t)(j0 + r_) * K_DH + c_;
            // IOFFSET applies to both LDS and global address (ISA 10.x async pseudocode)
            asm volatile("global_load_async_to_lds_b128 %0, %1, off"
                         :: "v"(ldsK), "v"(gk) : "memory");
            asm volatile("global_load_async_to_lds_b128 %0, %1, off offset:16"
                         :: "v"(ldsK), "v"(gk) : "memory");
            // --- V tile: load + transpose into LDS (needs VGPR pass anyway) ---
            const unsigned short* vs = Vb + (size_t)(j0 + r_) * K_DH + c_;
#pragma unroll
            for (int i = 0; i < 16; ++i) sVt[(c_ + i) * 40 + r_] = vs[i];
            // wave-local async completion, then block-wide visibility
            asm volatile("s_wait_asynccnt 0x0" ::: "memory");
        }
        __syncthreads();

        // S = Q Kt : two 16x16 subtiles (keys j0+0..15, j0+16..31), K-dim = d = 64
        v8f s0v = {}, s1v = {};
        FragBF bk;
        load_fragB(bk, sK + 0 * 72,       72, lane);  s0v = wmma_bf16(aq0, bk, s0v);
        load_fragB(bk, sK + 0 * 72 + 32,  72, lane);  s0v = wmma_bf16(aq1, bk, s0v);
        load_fragB(bk, sK + 16 * 72,      72, lane);  s1v = wmma_bf16(aq0, bk, s1v);
        load_fragB(bk, sK + 16 * 72 + 32, 72, lane);  s1v = wmma_bf16(aq1, bk, s1v);

        // online softmax (row m = v + 8*half lives on 16 lanes sharing `half`)
        float p0[8], p1[8];
#pragma unroll
        for (int v = 0; v < 8; ++v) {
            float mx = fmaxf(s0v[v], s1v[v]);
            mx = fmaxf(mx, __shfl_xor(mx, 1, 32));
            mx = fmaxf(mx, __shfl_xor(mx, 2, 32));
            mx = fmaxf(mx, __shfl_xor(mx, 4, 32));
            mx = fmaxf(mx, __shfl_xor(mx, 8, 32));
            float mn = fmaxf(mrow[v], mx);
            float scale = __expf(mrow[v] - mn);
            float e0 = __expf(s0v[v] - mn);
            float e1 = __expf(s1v[v] - mn);
            float rs = e0 + e1;
            rs += __shfl_xor(rs, 1, 32);
            rs += __shfl_xor(rs, 2, 32);
            rs += __shfl_xor(rs, 4, 32);
            rs += __shfl_xor(rs, 8, 32);
            lrow[v] = lrow[v] * scale + rs;
            mrow[v] = mn;
            o0[v] *= scale; o1[v] *= scale; o2[v] *= scale; o3[v] *= scale;
            p0[v] = e0; p1[v] = e1;
        }

        // stage P (16x32) into LDS to re-enter A-fragment layout
        unsigned short* pw_ = sP[wave];
#pragma unroll
        for (int v = 0; v < 8; ++v) {
            int m = v + 8 * half;
            pw_[m * 40 + n]      = f2bf(p0[v]);
            pw_[m * 40 + 16 + n] = f2bf(p1[v]);
        }
        __syncthreads();

        FragBF ap, bv;
        load_fragA(ap, pw_, 40, lane);
        load_fragB(bv, sVt + 0 * 16 * 40, 40, lane);  o0 = wmma_bf16(ap, bv, o0);
        load_fragB(bv, sVt + 1 * 16 * 40, 40, lane);  o1 = wmma_bf16(ap, bv, o1);
        load_fragB(bv, sVt + 2 * 16 * 40, 40, lane);  o2 = wmma_bf16(ap, bv, o2);
        load_fragB(bv, sVt + 3 * 16 * 40, 40, lane);  o3 = wmma_bf16(ap, bv, o3);
        __syncthreads();
    }

    // finalize: O[b][t][f], f = s*256 + h*64 + d
#pragma unroll
    for (int v = 0; v < 8; ++v) {
        float inv = 1.0f / lrow[v];
        int t = q0 + v + 8 * half;
        size_t rowbase = ((size_t)(b * K_T + t)) * 1024 + (size_t)s * 256 + h * 64;
        O[rowbase +  0 + n] = f2bf(o0[v] * inv);
        O[rowbase + 16 + n] = f2bf(o1[v] * inv);
        O[rowbase + 32 + n] = f2bf(o2[v] * inv);
        O[rowbase + 48 + n] = f2bf(o3[v] * inv);
    }
}

// ---------------- 6) output projection (WMMA, K=1024) ----------------
__global__ __launch_bounds__(128) void k_out(const unsigned short* __restrict__ O,
                                             const unsigned short* __restrict__ Wo,
                                             float* __restrict__ out) {
    int wave = threadIdx.x >> 5, lane = threadIdx.x & 31;
    int tile = blockIdx.x * 4 + wave;       // 288 row-tiles * 16 col-tiles = 4608
    int trow = tile >> 4;
    int tcol = tile & 15;
    const unsigned short* A  = O  + (size_t)trow * 16 * 1024;
    const unsigned short* Bw = Wo + (size_t)tcol * 16 * 1024;
    v8f acc = {};
    for (int k0 = 0; k0 < 1024; k0 += 32) {
        FragBF a, b;
        load_fragA(a, A + k0, 1024, lane);
        load_fragB(b, Bw + k0, 1024, lane);
        acc = wmma_bf16(a, b, acc);
    }
    int n = lane & 15, half = lane >> 4;
#pragma unroll
    for (int v = 0; v < 8; ++v) {
        int row = trow * 16 + v + 8 * half;  // = b*T + t
        out[(size_t)row * K_C + tcol * 16 + n] = acc[v];
    }
}

// ---------------- host launcher ----------------
extern "C" void kernel_launch(void* const* d_in, const int* in_sizes, int n_in,
                              void* d_out, int out_size, void* d_ws, size_t ws_size,
                              hipStream_t stream) {
    const float* x     = (const float*)d_in[0];
    const float* winw  = (const float*)d_in[1];
    const float* bn1g  = (const float*)d_in[2];
    const float* bn1b  = (const float*)d_in[3];
    const float* bn1m  = (const float*)d_in[4];
    const float* bn1v  = (const float*)d_in[5];
    const float* dww   = (const float*)d_in[6];
    const float* pww   = (const float*)d_in[7];
    const float* bn2g  = (const float*)d_in[8];
    const float* bn2b  = (const float*)d_in[9];
    const float* bn2m  = (const float*)d_in[10];
    const float* bn2v  = (const float*)d_in[11];
    const float* Wq    = (const float*)d_in[12];
    const float* Wk    = (const float*)d_in[13];
    const float* Wv    = (const float*)d_in[14];
    const float* Wout  = (const float*)d_in[15];
    (void)in_sizes; (void)n_in; (void)out_size; (void)ws_size;

    char* ws = (char*)d_ws;
    size_t off = 0;
    auto carve = [&](size_t bytes) { char* p = ws + off; off += (bytes + 255) & ~(size_t)255; return p; };

    float*          idn = (float*)carve((size_t)K_B * K_C * K_T * 4);          // 4.72 MB
    float*          ys  = (float*)carve((size_t)3 * K_B * K_C4 * K_T * 4);     // 3.54 MB
    unsigned short* r   = (unsigned short*)carve((size_t)4 * K_B * K_T * 64 * 2);
    unsigned short* WB  = (unsigned short*)carve((size_t)(3 * 16384 + 262144) * 2);
    unsigned short* Qb  = (unsigned short*)carve((size_t)K_SBH * K_T * K_DH * 2);
    unsigned short* Kb  = (unsigned short*)carve((size_t)K_SBH * K_T * K_DH * 2);
    unsigned short* Vb  = (unsigned short*)carve((size_t)K_SBH * K_T * K_DH * 2);
    unsigned short* Oc  = (unsigned short*)carve((size_t)K_B * K_T * 1024 * 2);

    // weights -> bf16
    k_cvt<<<(16384 + 255) / 256, 256, 0, stream>>>(Wq,   WB,          16384);
    k_cvt<<<(16384 + 255) / 256, 256, 0, stream>>>(Wk,   WB + 16384,  16384);
    k_cvt<<<(16384 + 255) / 256, 256, 0, stream>>>(Wv,   WB + 32768,  16384);
    k_cvt<<<(262144 + 255) / 256, 256, 0, stream>>>(Wout, WB + 49152, 262144);

    // conv_in + BN1
    k_convin<<<dim3(K_B * K_C, K_T / 256), 256, 0, stream>>>(x, winw, bn1g, bn1b, bn1m, bn1v, idn);

    // Res2Net scales (sequential dependency)
    float* y1 = ys;
    float* y2 = ys + (size_t)K_B * K_C4 * K_T;
    float* y3 = ys + (size_t)2 * K_B * K_C4 * K_T;
    k_res2<<<K_B * K_T, K_C4, 0, stream>>>(idn, idn, dww, pww, bn2g, bn2b, bn2m, bn2v, y1, 64, 0);
    k_res2<<<K_B * K_T, K_C4, 0, stream>>>(idn, y1,  dww, pww, bn2g, bn2b, bn2m, bn2v, y2, 128, 1);
    k_res2<<<K_B * K_T, K_C4, 0, stream>>>(idn, y2,  dww, pww, bn2g, bn2b, bn2m, bn2v, y3, 192, 1);

    // r bf16
    k_buildr<<<(4 * K_B * K_T * 64) / 256, 256, 0, stream>>>(idn, ys, r);

    // Q/K/V projections (WMMA)
    k_qkv<<<dim3(4 * K_B, (K_T / 16) * 16 / 4, 3), 128, 0, stream>>>(r, WB, Qb, Kb, Vb);

    // flash attention (WMMA + async LDS staging)
    k_attn<<<dim3(K_SBH, K_T / 64), 128, 0, stream>>>(Qb, Kb, Vb, Oc);

    // output projection (WMMA)
    k_out<<<(K_B * K_T / 16) * 16 / 4, 128, 0, stream>>>(Oc, WB + 49152, (float*)d_out);
}
